// Discriminator_35880156790839
// MI455X (gfx1250) — matrix operand order repaired and checked by reading, not compiled, not run on hardware
//
#include <hip/hip_runtime.h>

// ---------------------------------------------------------------------------
// Types for CDNA5 WMMA (wave32): v_wmma_f32_16x16x32_bf16
// ---------------------------------------------------------------------------
typedef __bf16 bf16_t;
typedef __attribute__((ext_vector_type(16))) __bf16 v16bf;
typedef __attribute__((ext_vector_type(8)))  float  v8f;

#define NCOLS 3072      // 3*H gate columns
#define HDIM  1024
#define EDIM  512
#define BSZ   64
#define SEQ   256

__device__ __forceinline__ float sigmoidf_(float x) {
    return 1.0f / (1.0f + __expf(-x));
}

// ---------------------------------------------------------------------------
// WMMA fragment layouts (bf16, 16x16x32, wave32):
//
// B fragment (K=32 x N=16), lane-major 16 halves/lane:
//   lane  0..15 : column n = nt*16 + lane,      K = kt*32 + {0..7, 16..23}
//   lane 16..31 : column n = nt*16 + (lane-16), K = kt*32 + {8..15, 24..31}
// B tiles stored GROUPED: tile(nt,kt) at ((((nt>>2)*Ktiles + kt)*4 + (nt&3)) * 512)
// halves, so one wave's 4 N-tiles for a given kt are contiguous (4096 bytes).
//
// A fragment (M=16 x K=32), tile (mt,kt) at ((mt*Ktiles+kt)*512) halves:
//   lane  0..15 : row m = mt*16+lane,    halves 0..7 -> K 0..7,  8..15 -> K 16..23
//   lane 16..31 : row m = mt*16+lane-16, halves 0..7 -> K 8..15, 8..15 -> K 24..31
// ---------------------------------------------------------------------------

// fp32 row-major W[3H, Kdim] -> bf16 grouped B-fragment tiles (B[k][n] = W[n][k])
__global__ void convert_weight_tiles(const float* __restrict__ W,
                                     bf16_t* __restrict__ out,
                                     int Kdim, int Ntiles) {
    int idx = blockIdx.x * blockDim.x + threadIdx.x;
    int Ktiles = Kdim >> 5;
    int total = Ntiles * Ktiles * 32;
    if (idx >= total) return;
    int lane = idx & 31;
    int tile = idx >> 5;                // storage-order tile id
    int nsub = tile & 3;
    int rest = tile >> 2;
    int kt   = rest % Ktiles;
    int ntg  = rest / Ktiles;
    int nt   = ntg * 4 + nsub;
    int n  = nt * 16 + (lane & 15);
    int kb = (kt << 5) + ((lane < 16) ? 0 : 8);
    bf16_t* o = out + ((size_t)tile << 9) + (lane << 4);
    const float* w = W + (size_t)n * Kdim + kb;
#pragma unroll
    for (int j = 0; j < 16; ++j) o[j] = (bf16_t)w[(j < 8) ? j : j + 8];
}

// X[B, S, E] fp32 -> per-step A-fragment tiles, all 256 steps at once.
// Tile id = t*64 + mt*16 + kt  (4 M-tiles, 16 K-tiles per step).
// Grid launched exactly: 256*4*16*32 threads.
__global__ void convert_x_tiles(const float* __restrict__ X, bf16_t* __restrict__ out) {
    int idx  = blockIdx.x * blockDim.x + threadIdx.x;
    int lane = idx & 31;
    int tile = idx >> 5;
    int kt = tile & 15;
    int mt = (tile >> 4) & 3;
    int t  = tile >> 6;
    int m  = mt * 16 + (lane & 15);
    int koff = (lane < 16) ? 0 : 8;
    const float* ap = X + ((size_t)m * SEQ + t) * EDIM + (kt << 5) + koff;
    bf16_t* o = out + ((size_t)tile << 9) + (lane << 4);
    float4 x0 = *(const float4*)(ap);
    float4 x1 = *(const float4*)(ap + 4);
    float4 x2 = *(const float4*)(ap + 16);
    float4 x3 = *(const float4*)(ap + 20);
    o[0]  = (bf16_t)x0.x; o[1]  = (bf16_t)x0.y; o[2]  = (bf16_t)x0.z; o[3]  = (bf16_t)x0.w;
    o[4]  = (bf16_t)x1.x; o[5]  = (bf16_t)x1.y; o[6]  = (bf16_t)x1.z; o[7]  = (bf16_t)x1.w;
    o[8]  = (bf16_t)x2.x; o[9]  = (bf16_t)x2.y; o[10] = (bf16_t)x2.z; o[11] = (bf16_t)x2.w;
    o[12] = (bf16_t)x3.x; o[13] = (bf16_t)x3.y; o[14] = (bf16_t)x3.z; o[15] = (bf16_t)x3.w;
}

// ---------------------------------------------------------------------------
// Dual GEMM: C[64, 3072] = A[64, K] x B[K, 3072], both operands pre-tiled bf16.
// Two independent problems selected by blockIdx.z (pi and ph of one GRU cell).
// block = 128 threads (4 waves); each wave owns one group of 4 N-tiles.
// Register double-buffered: iteration kt issues loads for kt+1, then runs the
// 4 v_wmma on kt's fragments, so WMMA never waits on the loads just issued.
// grid = (12, 4, 2).
// ---------------------------------------------------------------------------
__global__ void dual_gemm_wmma(const bf16_t* __restrict__ A0, const bf16_t* __restrict__ B0,
                               float* __restrict__ C0, int K0,
                               const bf16_t* __restrict__ A1, const bf16_t* __restrict__ B1,
                               float* __restrict__ C1, int K1) {
    const bf16_t* At; const bf16_t* Bt; float* C; int K;
    if (blockIdx.z == 0) { At = A0; Bt = B0; C = C0; K = K0; }
    else                 { At = A1; Bt = B1; C = C1; K = K1; }

    const int lane = threadIdx.x & 31;
    const int wave = threadIdx.x >> 5;
    const int mt   = blockIdx.y;
    const int ntg  = blockIdx.x * 4 + wave;   // N-tile group (4 tiles each)
    const int nk   = K >> 5;

    v8f acc[4];
#pragma unroll
    for (int n = 0; n < 4; ++n) acc[n] = {};

    const bf16_t* aptr = At + (((size_t)mt * nk) << 9) + (lane << 4);
    const bf16_t* bptr = Bt + (((size_t)ntg * nk) << 11) + (lane << 4);

    // Prologue: fragments for kt = 0
    v16bf af = *(const v16bf*)aptr;
    v16bf bf[4];
#pragma unroll
    for (int n = 0; n < 4; ++n) bf[n] = *(const v16bf*)(bptr + (n << 9));

    for (int kt = 0; kt < nk - 1; ++kt) {
        aptr += 512;
        bptr += 2048;
        __builtin_prefetch(bptr + 2048, 0, 3);   // K-group after next
        // Issue next iteration's loads first ...
        v16bf af_n = *(const v16bf*)aptr;
        v16bf bf_n[4];
#pragma unroll
        for (int n = 0; n < 4; ++n) bf_n[n] = *(const v16bf*)(bptr + (n << 9));
        // ... then compute on the current (already-arrived) fragments.
#pragma unroll
        for (int n = 0; n < 4; ++n) {
            acc[n] = __builtin_amdgcn_wmma_f32_16x16x32_bf16(false, af, false, bf[n],
                                                             (short)0, acc[n], false, false);
        }
        af = af_n;
#pragma unroll
        for (int n = 0; n < 4; ++n) bf[n] = bf_n[n];
    }
    // Epilogue: last K-step
#pragma unroll
    for (int n = 0; n < 4; ++n) {
        acc[n] = __builtin_amdgcn_wmma_f32_16x16x32_bf16(false, af, false, bf[n],
                                                         (short)0, acc[n], false, false);
    }

    // C/D layout: VGPR r -> row (r or 8+r), col = lane&15
    const int rowb = mt * 16 + ((lane < 16) ? 0 : 8);
    const int cb   = ntg * 64 + (lane & 15);
#pragma unroll
    for (int n = 0; n < 4; ++n) {
#pragma unroll
        for (int r = 0; r < 8; ++r) {
            C[(size_t)(rowb + r) * NCOLS + cb + n * 16] = acc[n][r];
        }
    }
}

// ---------------------------------------------------------------------------
// Block-wide LayerNorm over 1024 elems held as v[4] per thread (256 threads).
// ---------------------------------------------------------------------------
__device__ __forceinline__ void layernorm4(float v[4], const float* __restrict__ g,
                                           const float* __restrict__ b, int base,
                                           float* sh) {
    float s = 0.f, q = 0.f;
#pragma unroll
    for (int i = 0; i < 4; ++i) { s += v[i]; q += v[i] * v[i]; }
    int tid = threadIdx.x;
    sh[tid] = s; sh[256 + tid] = q;
    __syncthreads();
    for (int off = 128; off > 0; off >>= 1) {
        if (tid < off) { sh[tid] += sh[tid + off]; sh[256 + tid] += sh[256 + tid + off]; }
        __syncthreads();
    }
    float mean = sh[0] * (1.0f / 1024.0f);
    float var  = sh[256] * (1.0f / 1024.0f) - mean * mean;
    float inv  = rsqrtf(var + 1e-5f);
    __syncthreads();
#pragma unroll
    for (int i = 0; i < 4; ++i) {
        int j = tid + i * 256;
        v[i] = (v[i] - mean) * inv * g[base + j] + b[base + j];
    }
}

// ---------------------------------------------------------------------------
// GRU pointwise: bias + per-gate LayerNorm on pi and ph, gates, h update.
// Writes both fp32 h and the bf16 A-fragment tiled copy consumed by the next
// WMMA GEMM. One block (256 thr) per batch row.
// ---------------------------------------------------------------------------
__global__ void gru_pointwise(const float* __restrict__ pi, const float* __restrict__ ph,
                              const float* __restrict__ bih, const float* __restrict__ bhh,
                              const float* __restrict__ gig, const float* __restrict__ gib,
                              const float* __restrict__ ghg, const float* __restrict__ ghb,
                              float* __restrict__ h, bf16_t* __restrict__ hA) {
    __shared__ float sh[512];
    const int b = blockIdx.x;
    const int tid = threadIdx.x;
    float vi[3][4], vh[3][4];
#pragma unroll
    for (int g = 0; g < 3; ++g) {
#pragma unroll
        for (int i = 0; i < 4; ++i) {
            int jl = tid + i * 256;
            int jg = g * HDIM + jl;
            vi[g][i] = pi[(size_t)b * NCOLS + jg] + bih[jg];
            vh[g][i] = ph[(size_t)b * NCOLS + jg] + bhh[jg];
        }
    }
    for (int g = 0; g < 3; ++g) {
        layernorm4(vi[g], gig, gib, g * HDIM, sh);
        layernorm4(vh[g], ghg, ghb, g * HDIM, sh);
    }
#pragma unroll
    for (int i = 0; i < 4; ++i) {
        int j = tid + i * 256;
        float r = sigmoidf_(vi[0][i] + vh[0][i]);
        float z = sigmoidf_(vi[1][i] + vh[1][i]);
        float n = tanhf(vi[2][i] + r * vh[2][i]);
        float hp = h[(size_t)b * HDIM + j];
        float hn = (1.0f - z) * n + z * hp;
        h[(size_t)b * HDIM + j] = hn;
        // scatter into A-fragment tiled layout: m=b, k=j
        int kt  = j >> 5;
        int kin = j & 31;
        int sub = kin & 7;
        int grp = kin >> 3;                       // 0..3
        int lane = ((grp & 1) ? 16 : 0) + (b & 15);
        int half = ((grp >> 1) ? 8 : 0) + sub;
        hA[(((size_t)(b >> 4) * 32 + kt) << 9) + lane * 16 + half] = (bf16_t)hn;
    }
}

// ---------------------------------------------------------------------------
__global__ void zero_f32(float* p, int n) {
    int i = blockIdx.x * blockDim.x + threadIdx.x;
    if (i < n) p[i] = 0.0f;
}

__global__ void score_kernel(const float* __restrict__ h, const float* __restrict__ Wout,
                             const float* __restrict__ bout, float* __restrict__ sc) {
    __shared__ float sh[256];
    const int b = blockIdx.x, tid = threadIdx.x;
    float s = 0.f;
#pragma unroll
    for (int i = 0; i < 4; ++i) {
        int j = tid + i * 256;
        s += h[(size_t)b * HDIM + j] * Wout[j];
    }
    sh[tid] = s; __syncthreads();
    for (int off = 128; off > 0; off >>= 1) {
        if (tid < off) sh[tid] += sh[tid + off];
        __syncthreads();
    }
    if (tid == 0) sc[b] = sh[0] + bout[0];
}

__global__ void finalize_kernel(const float* __restrict__ st, const float* __restrict__ sg,
                                float* __restrict__ out) {
    __shared__ float sh[128];
    int tid = threadIdx.x; // 64 threads
    sh[tid] = st[tid]; sh[64 + tid] = sg[tid];
    __syncthreads();
    for (int off = 32; off > 0; off >>= 1) {
        if (tid < off) { sh[tid] += sh[tid + off]; sh[64 + tid] += sh[64 + tid + off]; }
        __syncthreads();
    }
    if (tid == 0) {
        float mt = sh[0] * (1.0f / 64.0f);
        float mg = sh[64] * (1.0f / 64.0f);
        out[0] = -mt + mg;   // discriminator loss
        out[1] = -mg;        // generator loss
    }
}

// ---------------------------------------------------------------------------
extern "C" void kernel_launch(void* const* d_in, const int* in_sizes, int n_in,
                              void* d_out, int out_size, void* d_ws, size_t ws_size,
                              hipStream_t stream) {
    const float* gen  = (const float*)d_in[0];
    const float* tru  = (const float*)d_in[1];
    const float* Wih[2] = { (const float*)d_in[2],  (const float*)d_in[10] };
    const float* Whh[2] = { (const float*)d_in[3],  (const float*)d_in[11] };
    const float* bih[2] = { (const float*)d_in[4],  (const float*)d_in[12] };
    const float* bhh[2] = { (const float*)d_in[5],  (const float*)d_in[13] };
    const float* gig[2] = { (const float*)d_in[6],  (const float*)d_in[14] };
    const float* gib[2] = { (const float*)d_in[7],  (const float*)d_in[15] };
    const float* ghg[2] = { (const float*)d_in[8],  (const float*)d_in[16] };
    const float* ghb[2] = { (const float*)d_in[9],  (const float*)d_in[17] };
    const float* Wout = (const float*)d_in[18];
    const float* bout = (const float*)d_in[19];

    // Workspace carve-up (~42 MB)
    char* ws = (char*)d_ws;
    auto carve = [&](size_t bytes) {
        void* p = (void*)ws;
        ws += (bytes + 255) & ~(size_t)255;
        return p;
    };
    const int Ntiles = NCOLS / 16;                 // 192
    bf16_t* WihT[2]; bf16_t* WhhT[2];
    WihT[0] = (bf16_t*)carve((size_t)Ntiles * (EDIM / 32) * 512 * sizeof(bf16_t));
    WhhT[0] = (bf16_t*)carve((size_t)Ntiles * (HDIM / 32) * 512 * sizeof(bf16_t));
    WihT[1] = (bf16_t*)carve((size_t)Ntiles * (HDIM / 32) * 512 * sizeof(bf16_t));
    WhhT[1] = (bf16_t*)carve((size_t)Ntiles * (HDIM / 32) * 512 * sizeof(bf16_t));
    bf16_t* XA  = (bf16_t*)carve((size_t)SEQ * 64 * 512 * sizeof(bf16_t)); // 256 steps x (4 mt x 16 kt) tiles
    bf16_t* h0A = (bf16_t*)carve((size_t)BSZ * HDIM * sizeof(bf16_t));
    bf16_t* h1A = (bf16_t*)carve((size_t)BSZ * HDIM * sizeof(bf16_t));
    float* pi0 = (float*)carve((size_t)BSZ * NCOLS * sizeof(float));
    float* ph0 = (float*)carve((size_t)BSZ * NCOLS * sizeof(float));
    float* pi1 = (float*)carve((size_t)BSZ * NCOLS * sizeof(float));
    float* ph1 = (float*)carve((size_t)BSZ * NCOLS * sizeof(float));
    float* h0  = (float*)carve((size_t)BSZ * HDIM * sizeof(float));
    float* h1  = (float*)carve((size_t)BSZ * HDIM * sizeof(float));
    float* s_true = (float*)carve(BSZ * sizeof(float));
    float* s_gen  = (float*)carve(BSZ * sizeof(float));

    // Weight conversion (every call: deterministic, no caching)
    {
        int t0 = Ntiles * (EDIM / 32) * 32;    // Wih0
        int t1 = Ntiles * (HDIM / 32) * 32;    // Whh0 / Wih1 / Whh1
        convert_weight_tiles<<<(t0 + 255) / 256, 256, 0, stream>>>(Wih[0], WihT[0], EDIM, Ntiles);
        convert_weight_tiles<<<(t1 + 255) / 256, 256, 0, stream>>>(Whh[0], WhhT[0], HDIM, Ntiles);
        convert_weight_tiles<<<(t1 + 255) / 256, 256, 0, stream>>>(Wih[1], WihT[1], HDIM, Ntiles);
        convert_weight_tiles<<<(t1 + 255) / 256, 256, 0, stream>>>(Whh[1], WhhT[1], HDIM, Ntiles);
    }

    const dim3 ggrid(NCOLS / 256, BSZ / 16, 2);   // (12, 4, 2)
    const dim3 gblock(128);
    const int  xconv_blocks = (SEQ * 64 * 32) / 256;  // exact

    for (int s = 0; s < 2; ++s) {
        const float* X = (s == 0) ? tru : gen;
        float* sc = (s == 0) ? s_true : s_gen;

        // Pre-tile the whole input sequence into bf16 A-fragments (one pass).
        convert_x_tiles<<<xconv_blocks, 256, 0, stream>>>(X, XA);

        zero_f32<<<(BSZ * HDIM + 255) / 256, 256, 0, stream>>>(h0, BSZ * HDIM);
        zero_f32<<<(BSZ * HDIM + 255) / 256, 256, 0, stream>>>(h1, BSZ * HDIM);
        // bf16 zero has all-zero bits: clear the fragment buffers as floats
        zero_f32<<<(BSZ * HDIM / 2 + 255) / 256, 256, 0, stream>>>((float*)h0A, BSZ * HDIM / 2);
        zero_f32<<<(BSZ * HDIM / 2 + 255) / 256, 256, 0, stream>>>((float*)h1A, BSZ * HDIM / 2);

        for (int t = 0; t < SEQ; ++t) {
            // Layer 0: pi0 = x_t @ Wih0^T, ph0 = h0 @ Whh0^T
            dual_gemm_wmma<<<ggrid, gblock, 0, stream>>>(
                XA + (size_t)t * 64 * 512, WihT[0], pi0, EDIM,
                h0A, WhhT[0], ph0, HDIM);
            gru_pointwise<<<BSZ, 256, 0, stream>>>(pi0, ph0, bih[0], bhh[0],
                                                   gig[0], gib[0], ghg[0], ghb[0], h0, h0A);
            // Layer 1: pi1 = h0(new) @ Wih1^T, ph1 = h1 @ Whh1^T
            dual_gemm_wmma<<<ggrid, gblock, 0, stream>>>(
                h0A, WihT[1], pi1, HDIM,
                h1A, WhhT[1], ph1, HDIM);
            gru_pointwise<<<BSZ, 256, 0, stream>>>(pi1, ph1, bih[1], bhh[1],
                                                   gig[1], gib[1], ghg[1], ghb[1], h1, h1A);
        }
        score_kernel<<<BSZ, 256, 0, stream>>>(h1, Wout, bout, sc);
    }

    finalize_kernel<<<1, 64, 0, stream>>>(s_true, s_gen, (float*)d_out);
}
